// QRNNClassifier_16449724745411
// MI455X (gfx1250) — matrix operand order repaired and checked by reading, not compiled, not run on hardware
//
#include <hip/hip_runtime.h>
#include <math.h>
#include <stdint.h>

#define T_   256
#define B_   128
#define EMB_ 1024
#define HID_ 1024
#define H3_  3072
#define LAB_ 50

// ---- LDS layout (in halves) for the fused kernel ----
#define BROW_        1032                 // padded B row stride (1024 + 8)
#define AROW_        40                   // padded A row stride (32 + 8)
#define BS_HALVES_   (96 * BROW_)         // z|f|o weights, 96 rows -> 198,144 B
#define ABUF_HALVES_ (64 * AROW_)         // one A chunk buffer: 64 rows
#define AS_HALVES_   (4 * ABUF_HALVES_)   // 4-deep ring            20,480 B
#define LDS_BYTES_   ((BS_HALVES_ + AS_HALVES_) * 2)   // ~218 KB of 320 KB
#define NCHUNK_MAIN_ (T_ * 32)            // 8192 K-chunks in the t-loop
#define NCHUNK_ALL_  (NCHUNK_MAIN_ + 32)  // + 32 o-gate chunks

typedef __attribute__((ext_vector_type(16))) _Float16 v16h;
typedef __attribute__((ext_vector_type(8)))  _Float16 v8h;
typedef __attribute__((ext_vector_type(4)))  _Float16 v4h;
typedef __attribute__((ext_vector_type(8)))  float    v8f;

// ---- fast transcendental helpers (single v_exp_f32 / v_rcp_f32 ops; no
// libm branchy expansions inside the sequential scan) ----
#if __has_builtin(__builtin_amdgcn_exp2f)
#define EXP2F(x) __builtin_amdgcn_exp2f(x)
#else
#define EXP2F(x) exp2f(x)
#endif
#if __has_builtin(__builtin_amdgcn_rcpf)
#define RCPF(x) __builtin_amdgcn_rcpf(x)
#else
#define RCPF(x) (1.0f / (x))
#endif
#if __has_builtin(__builtin_amdgcn_logf)
#define LOG2F_(x) __builtin_amdgcn_logf(x)
#else
#define LOG2F_(x) log2f(x)
#endif
#define LOG2E_ 1.442695040888963f
#define LN2_   0.693147180559945f

__device__ __forceinline__ float fast_sigmoid(float x)
{
    return RCPF(1.0f + EXP2F(-LOG2E_ * x));
}
__device__ __forceinline__ float fast_tanh(float x)
{
#if __has_builtin(__builtin_amdgcn_tanhf)
    return __builtin_amdgcn_tanhf(x);
#else
    // tanh(x) = 1 - 2/(1 + e^{2x})
    return 1.0f - 2.0f * RCPF(1.0f + EXP2F((2.0f * LOG2E_) * x));
#endif
}

// CDNA5 async copy: memory -> LDS, no VGPR round trip, tracked by ASYNCcnt.
__device__ __forceinline__ void async_b128(uint32_t lds_addr, uint64_t gaddr)
{
    asm volatile("global_load_async_to_lds_b128 %0, %1, off"
                 :: "v"(lds_addr), "v"(gaddr) : "memory");
}

// ---------------------------------------------------------------------------
// Kernel 1: W [EMB,3H] f32 -> WhT [3H,EMB] f16 (transposed: B tiles contiguous
// along K, matching the WMMA B-fragment layout).
// ---------------------------------------------------------------------------
__global__ __launch_bounds__(256) void k_convert_w(const float* __restrict__ W,
                                                   _Float16* __restrict__ WhT)
{
    size_t idx = (size_t)blockIdx.x * 256 + threadIdx.x;
    if (idx >= (size_t)EMB_ * H3_) return;
    int k = (int)(idx / H3_);
    int n = (int)(idx % H3_);
    WhT[(size_t)n * EMB_ + k] = (_Float16)W[idx];
}

// ---------------------------------------------------------------------------
// Kernel 2: embedding gather + f32->f16 convert into Xh[T*B][EMB].
// ---------------------------------------------------------------------------
__global__ __launch_bounds__(256) void k_gather(const int* __restrict__ sent,
                                                const float* __restrict__ emb,
                                                _Float16* __restrict__ Xh)
{
    int row = blockIdx.x;               // t*B + b
    int tok = sent[row];
    const float4 v = ((const float4*)(emb + (size_t)tok * EMB_))[threadIdx.x];
    v4h h;
    h[0] = (_Float16)v.x; h[1] = (_Float16)v.y;
    h[2] = (_Float16)v.z; h[3] = (_Float16)v.w;
    *(v4h*)(Xh + (size_t)row * EMB_ + threadIdx.x * 4) = h;
}

// ---------------------------------------------------------------------------
// Kernel 3: fused gate GEMM (WMMA f16->f32) + fo-pool scan.
// WG tile M=64 x N=32, 8 waves; wave (mw,nw) owns one 16x16 tile, keeps its
// c-fragment in registers over all 256 timesteps.
//   * all z|f|o weight rows for this N-tile resident in LDS (loaded once,
//     async), ~194 KB of the WGP's 320 KB.
//   * A chunks stream through a 4-deep LDS ring via global_load_async_to_lds,
//     issue distance 2, one barrier per chunk, s_wait_asynccnt pacing.
// ---------------------------------------------------------------------------
__global__ __launch_bounds__(256) void k_qrnn(const _Float16* __restrict__ Xh,
                                              const _Float16* __restrict__ WhT,
                                              const float* __restrict__ bias,
                                              float* __restrict__ h_last)
{
    extern __shared__ _Float16 smem[];
    _Float16* Bs = smem;                  // [96][BROW_]  (rows: g*32 + n)
    _Float16* As = smem + BS_HALVES_;     // [4][64][AROW_]

    const int tid  = threadIdx.x;
    const int lane = tid & 31;
    const int wave = tid >> 5;
    const int mw   = wave & 3;
    const int nw   = wave >> 2;
    const int N0   = blockIdx.x * 32;     // hid column base
    const int M0   = blockIdx.y * 64;     // batch row base

    const uint32_t ldsBs = (uint32_t)(uintptr_t)Bs;
    const uint32_t ldsAs = (uint32_t)(uintptr_t)As;

    // cooperative A staging indices (256 threads x one b128 per chunk)
    const int ar  = tid >> 2;             // 0..63 : A row
    const int ac8 = (tid & 3) * 8;        // K-half offset

    // 1) Resident weights: 96 rows x 1024 halves, async -> LDS (48 per thread)
    for (int j = tid; j < 96 * 128; j += 256) {
        int row = j >> 7;                 // 0..95  (= g*32 + n)
        int c8  = (j & 127) * 8;
        int g   = row >> 5, n = row & 31;
        uint64_t ga = (uint64_t)(uintptr_t)
            (WhT + ((size_t)g * HID_ + N0 + n) * EMB_ + c8);
        async_b128(ldsBs + (uint32_t)(row * BROW_ + c8) * 2, ga);
    }

    // flattened A-chunk issue: jj in [0, NCHUNK_ALL_); o-pass reuses t = T-1
    auto issueA = [&](int jj) {
        if (jj >= NCHUNK_ALL_) return;
        int tt = (jj < NCHUNK_MAIN_) ? (jj >> 5) : (T_ - 1);
        int kk = (jj & 31) * 32;
        int bu = jj & 3;
        uint64_t ga = (uint64_t)(uintptr_t)
            (Xh + ((size_t)tt * B_ + M0 + ar) * EMB_ + kk + ac8);
        async_b128(ldsAs + (uint32_t)(bu * ABUF_HALVES_ + ar * AROW_ + ac8) * 2, ga);
    };
    issueA(0);
    issueA(1);

    // WMMA fragment indices (contiguous k = 16*(lane/16) + s in both A and B)
    const int am   = mw * 16 + (lane & 15);       // A: lane -> row m
    const int ab0  = (lane >> 4) * 16;            // 16 consecutive K halves
    const int bn   = nw * 16 + (lane & 15);       // B: lane -> column n
    const int bk   = (lane >> 4) * 16;            // 16 consecutive K halves
    const int ncol = N0 + nw * 16 + (lane & 15);  // C: lane -> N

    const float bz = bias[ncol];
    const float bf = bias[HID_ + ncol];
    const float bo = bias[2 * HID_ + ncol];

    v8f c = {};                                   // fo-pool state, c_0 = 0

    for (int t = 0; t < T_; ++t) {
        v8f az = {}, af = {};
        for (int i = 0; i < 32; ++i) {
            issueA(t * 32 + i + 2);               // distance-2 prefetch
            asm volatile("s_wait_asynccnt 0x2" ::: "memory");
            __syncthreads();                      // single barrier per chunk
            const _Float16* ab = As + (i & 3) * ABUF_HALVES_;
            v8h alo = *(const v8h*)(ab + am * AROW_ + ab0);
            v8h ahi = *(const v8h*)(ab + am * AROW_ + ab0 + 8);
            v16h a  = __builtin_shufflevector(alo, ahi,
                        0,1,2,3,4,5,6,7,8,9,10,11,12,13,14,15);
            const int kf = i * 32 + bk;
            v8h z0 = *(const v8h*)(Bs + bn * BROW_ + kf);
            v8h z1 = *(const v8h*)(Bs + bn * BROW_ + kf + 8);
            v8h f0 = *(const v8h*)(Bs + (32 + bn) * BROW_ + kf);
            v8h f1 = *(const v8h*)(Bs + (32 + bn) * BROW_ + kf + 8);
            v16h vbz = __builtin_shufflevector(z0, z1,
                        0,1,2,3,4,5,6,7,8,9,10,11,12,13,14,15);
            v16h vbf = __builtin_shufflevector(f0, f1,
                        0,1,2,3,4,5,6,7,8,9,10,11,12,13,14,15);
            az = __builtin_amdgcn_wmma_f32_16x16x32_f16(false, a, false, vbz,
                                                        (short)0, az, false, false);
            af = __builtin_amdgcn_wmma_f32_16x16x32_f16(false, a, false, vbf,
                                                        (short)0, af, false, false);
        }
        #pragma unroll
        for (int i = 0; i < 8; ++i) {
            float z = fast_tanh(az[i] + bz);
            float f = fast_sigmoid(af[i] + bf);
            c[i] = f * z + (1.0f - f) * c[i];
        }
    }

    // o-gate pass over the final timestep (A chunks continue the same ring)
    v8f ao = {};
    for (int i = 0; i < 32; ++i) {
        if (i < 30) {
            issueA(NCHUNK_MAIN_ + i + 2);
            asm volatile("s_wait_asynccnt 0x2" ::: "memory");
        } else if (i == 30) {
            asm volatile("s_wait_asynccnt 0x1" ::: "memory");
        } else {
            asm volatile("s_wait_asynccnt 0x0" ::: "memory");
        }
        __syncthreads();
        const _Float16* ab = As + (i & 3) * ABUF_HALVES_;
        v8h alo = *(const v8h*)(ab + am * AROW_ + ab0);
        v8h ahi = *(const v8h*)(ab + am * AROW_ + ab0 + 8);
        v16h a  = __builtin_shufflevector(alo, ahi,
                    0,1,2,3,4,5,6,7,8,9,10,11,12,13,14,15);
        const int kf = i * 32 + bk;
        v8h o0 = *(const v8h*)(Bs + (64 + bn) * BROW_ + kf);
        v8h o1 = *(const v8h*)(Bs + (64 + bn) * BROW_ + kf + 8);
        v16h vbo = __builtin_shufflevector(o0, o1,
                    0,1,2,3,4,5,6,7,8,9,10,11,12,13,14,15);
        ao = __builtin_amdgcn_wmma_f32_16x16x32_f16(false, a, false, vbo,
                                                    (short)0, ao, false, false);
    }
    #pragma unroll
    for (int i = 0; i < 8; ++i) {
        float o = fast_sigmoid(ao[i] + bo);
        int mrow = M0 + mw * 16 + i + 8 * (lane >> 4);  // C: M = vgpr + 8*(lane>=16)
        h_last[(size_t)mrow * HID_ + ncol] = o * c[i];
    }
}

// ---------------------------------------------------------------------------
// Kernel 4: logits = h_last @ Wout + bout ; log_softmax over 50 labels.
// 256 threads = 64 labels x 4 K-segments; LDS partial reduction.
// ---------------------------------------------------------------------------
__global__ __launch_bounds__(256) void k_head(const float* __restrict__ h_last,
                                              const float* __restrict__ Wout,
                                              const float* __restrict__ bout,
                                              float* __restrict__ out)
{
    __shared__ float sh[HID_];
    __shared__ float part[64][4];
    __shared__ float slog[LAB_];
    __shared__ float sred[2];
    const int b   = blockIdx.x;
    const int tid = threadIdx.x;
    for (int i = tid; i < HID_; i += 256)
        sh[i] = h_last[(size_t)b * HID_ + i];
    __syncthreads();
    const int lab = tid >> 2;
    const int seg = tid & 3;
    if (lab < LAB_) {
        float acc = 0.0f;
        const int k0 = seg * 256;
        for (int k = 0; k < 256; ++k)
            acc = fmaf(sh[k0 + k], Wout[(size_t)(k0 + k) * LAB_ + lab], acc);
        part[lab][seg] = acc;
    }
    __syncthreads();
    if (tid < LAB_)
        slog[tid] = bout[tid] + part[tid][0] + part[tid][1]
                              + part[tid][2] + part[tid][3];
    __syncthreads();
    if (tid == 0) {
        float m = slog[0];
        for (int j = 1; j < LAB_; ++j) m = fmaxf(m, slog[j]);
        float s = 0.0f;
        for (int j = 0; j < LAB_; ++j) s += EXP2F((slog[j] - m) * LOG2E_);
        sred[0] = m;
        sred[1] = LOG2F_(s) * LN2_;
    }
    __syncthreads();
    if (tid < LAB_)
        out[(size_t)b * LAB_ + tid] = slog[tid] - sred[0] - sred[1];
}

// ---------------------------------------------------------------------------
extern "C" void kernel_launch(void* const* d_in, const int* in_sizes, int n_in,
                              void* d_out, int out_size, void* d_ws, size_t ws_size,
                              hipStream_t stream)
{
    (void)in_sizes; (void)n_in; (void)out_size; (void)ws_size;
    const int*   sent = (const int*)d_in[0];
    const float* emb  = (const float*)d_in[1];
    const float* W    = (const float*)d_in[2];
    const float* bias = (const float*)d_in[3];
    const float* Wout = (const float*)d_in[4];
    const float* bout = (const float*)d_in[5];

    // workspace: Xh (64 MB f16) | WhT (6 MB f16) | h_last (0.5 MB f32)
    const size_t xh_bytes  = (size_t)T_ * B_ * EMB_ * sizeof(_Float16);
    const size_t wht_bytes = (size_t)EMB_ * H3_ * sizeof(_Float16);
    _Float16* Xh    = (_Float16*)d_ws;
    _Float16* WhT   = (_Float16*)((char*)d_ws + xh_bytes);
    float*    hlast = (float*)((char*)d_ws + xh_bytes + wht_bytes);

    k_convert_w<<<(EMB_ * H3_) / 256, 256, 0, stream>>>(W, WhT);
    k_gather<<<T_ * B_, 256, 0, stream>>>(sent, emb, Xh);
    k_qrnn<<<dim3(HID_ / 32, B_ / 64), 256, LDS_BYTES_, stream>>>(Xh, WhT, bias, hlast);
    k_head<<<B_, 256, 0, stream>>>(hlast, Wout, bout, (float*)d_out);
}
